// MetaMultiHeadedAttention_1846835938037
// MI455X (gfx1250) — compile-verified, matrix operand order checked by reading, not run on hardware
//
#include <hip/hip_runtime.h>

// MI455X / gfx1250 multi-head attention, f16 WMMA pipeline with TDM staging.
// B=4, S=2048, D=1024, H=16, DK=64.
// Workspace layout (bytes):
//   [0,16M) Xq f16   [16M,32M) Xk   [32M,48M) Xv          (activations, f16)
//   [48M,50M) Wq f16 [50M,52M) Wk [52M,54M) Wv [54M,56M) Wo
//   [56M,72M) Q [B,H,S,DK]  [72M,88M) K [B,H,S,DK]
//   [88M,104M) V^T [B,H,DK,S]  [104M,120M) Ctx [B,S,D]

typedef __attribute__((ext_vector_type(16))) _Float16 v16h;
typedef __attribute__((ext_vector_type(8)))  _Float16 v8h;
typedef __attribute__((ext_vector_type(8)))  float    v8f;
typedef __attribute__((ext_vector_type(4)))  unsigned int u32x4;
typedef __attribute__((ext_vector_type(4)))  int i32x4;
typedef __attribute__((ext_vector_type(8)))  int i32x8;

__device__ __forceinline__ v8f wmma16(v16h a, v16h b, v8f c) {
  return __builtin_amdgcn_wmma_f32_16x16x32_f16(
      false, a, false, b, (short)0, c, false, false);
}

// 16x32 f16 A/B fragment from row-major memory (global or LDS after inlining).
// lane L: row L&15; lanes 0-15 hold K[0..7]/[16..23], lanes 16-31 K[8..15]/[24..31].
__device__ __forceinline__ v16h load_frag(const _Float16* base, int ld) {
  const int lane = threadIdx.x & 31;
  const int r  = lane & 15;
  const int kh = (lane >> 4) << 3;
  const _Float16* p = base + (size_t)r * ld + kh;
  v8h lo = *(const v8h*)(p);
  v8h hi = *(const v8h*)(p + 16);
  return __builtin_shufflevector(lo, hi, 0,1,2,3,4,5,6,7,8,9,10,11,12,13,14,15);
}

// Issue a 2D TDM tile load: global (row-major, element=2B) -> LDS.
// D# packing per CDNA5 ISA 08_async_tensor §8.3/8.4. Dims/strides in elements.
// This toolchain's builtin is the 6-arg form:
//   (u32x4 g0, i32x8 g1, i32x4 g2, i32x4 g3, i32x8 g4, i32 cpol)
__device__ __forceinline__ void tdm_load_2d(unsigned lds_off, const void* gaddr,
                                            unsigned td0, unsigned td1,
                                            unsigned tl0, unsigned tl1,
                                            unsigned long long stride0) {
  unsigned long long ga = (unsigned long long)(uintptr_t)gaddr;
  u32x4 g0;
  g0[0] = 1u;                                    // count=1, user descriptor
  g0[1] = lds_off;                               // lds_addr (bytes)
  g0[2] = (unsigned)ga;                          // global_addr[31:0]
  g0[3] = (unsigned)((ga >> 32) & 0x01ffffffu)   // global_addr[56:32]
          | (2u << 30);                          // type = 2 ("image")
  i32x8 g1;
  g1[0] = (int)(1u << 16);                       // data_size=1 (2B), wg_mask=0
  g1[1] = (int)((td0 & 0xffffu) << 16);          // tensor_dim0[15:0] @ [63:48]
  g1[2] = (int)(((td0 >> 16) & 0xffffu) | ((td1 & 0xffffu) << 16));
  g1[3] = (int)(((td1 >> 16) & 0xffffu) | ((tl0 & 0xffffu) << 16));  // tile_dim0
  g1[4] = (int)(tl1 & 0xffffu);                  // tile_dim1, tile_dim2=0
  g1[5] = (int)(stride0 & 0xffffffffu);          // tensor_dim0_stride[31:0]
  g1[6] = (int)((stride0 >> 32) & 0xffffu);      // stride[47:32], dim1_stride=0
  g1[7] = 0;
  i32x4 z4; z4[0] = z4[1] = z4[2] = z4[3] = 0;
  i32x8 z8;
  #pragma unroll
  for (int i = 0; i < 8; i++) z8[i] = 0;
  __builtin_amdgcn_tensor_load_to_lds(g0, g1, z4, z4, z8, 0);
}

__device__ __forceinline__ unsigned lds_offset(const void* p) {
  return (unsigned)(uintptr_t)p;
}

__global__ void cvt_f32_f16_kernel(const float* __restrict__ s,
                                   _Float16* __restrict__ d, int n) {
  int i = blockIdx.x * blockDim.x + threadIdx.x;
  if (i < n) d[i] = (_Float16)s[i];
}

// 128x128 block GEMM, K staged 32 at a time via TDM into LDS, double buffered.
// 256 threads = 8 waves (4 M x 2 N), each wave 32x64 = 2x4 WMMA tiles.
// MODE 0: f16 [B,H,S,DK]; MODE 1: f16 [B,H,DK,S]; MODE 2: f32 row-major.
template<int MODE>
__global__ __launch_bounds__(256)
void gemm_wmma_kernel(const _Float16* __restrict__ A,
                      const _Float16* __restrict__ W,
                      const float*    __restrict__ bias,
                      void* __restrict__ out) {
  constexpr int Kd = 1024;
  constexpr int KT = Kd / 32;
  __shared__ __align__(16) _Float16 Asm[2][128 * 32];
  __shared__ __align__(16) _Float16 Bsm[2][128 * 32];

  const int wid  = threadIdx.x >> 5;
  const int lane = threadIdx.x & 31;
  const int rb   = blockIdx.y * 128;
  const int cb   = blockIdx.x * 128;
  const int row0 = rb + (wid & 3) * 32;
  const int col0 = cb + (wid >> 2) * 64;

  v8f acc[2][4] = {};

  if (threadIdx.x < 32) {   // wave 0 drives the TDM
    tdm_load_2d(lds_offset(&Asm[0][0]), A + (size_t)rb * Kd, Kd, 8192, 32, 128, Kd);
    tdm_load_2d(lds_offset(&Bsm[0][0]), W + (size_t)cb * Kd, Kd, 1024, 32, 128, Kd);
  }

  for (int t = 0; t < KT; ++t) {
    const int buf = t & 1;
    if (threadIdx.x < 32) {
      if (t + 1 < KT) {
        const int k1 = (t + 1) * 32;
        tdm_load_2d(lds_offset(&Asm[buf ^ 1][0]), A + (size_t)rb * Kd + k1,
                    Kd, 8192, 32, 128, Kd);
        tdm_load_2d(lds_offset(&Bsm[buf ^ 1][0]), W + (size_t)cb * Kd + k1,
                    Kd, 1024, 32, 128, Kd);
        __builtin_amdgcn_s_wait_tensorcnt(2);   // tile t landed, t+1 in flight
      } else {
        __builtin_amdgcn_s_wait_tensorcnt(0);
      }
    }
    __syncthreads();

    v16h af[2], bf[4];
    #pragma unroll
    for (int i = 0; i < 2; i++)
      af[i] = load_frag(&Asm[buf][((wid & 3) * 32 + i * 16) * 32], 32);
    #pragma unroll
    for (int j = 0; j < 4; j++)
      bf[j] = load_frag(&Bsm[buf][((wid >> 2) * 64 + j * 16) * 32], 32);
    #pragma unroll
    for (int i = 0; i < 2; i++)
      #pragma unroll
      for (int j = 0; j < 4; j++)
        acc[i][j] = wmma16(af[i], bf[j], acc[i][j]);

    __syncthreads();   // reads done before this buffer is refilled at t+2
  }

  const int hs = lane >> 4, nl = lane & 15;
  #pragma unroll
  for (int i = 0; i < 2; i++) {
    #pragma unroll
    for (int j = 0; j < 4; j++) {
      #pragma unroll
      for (int r = 0; r < 8; r++) {
        int m = row0 + i * 16 + hs * 8 + r;
        int n = col0 + j * 16 + nl;
        float v = acc[i][j][r] + bias[n];
        if (MODE == 0) {
          int b = m >> 11, s = m & 2047, h = n >> 6, dk = n & 63;
          ((_Float16*)out)[(((size_t)(b * 16 + h) * 2048 + s) << 6) + dk] = (_Float16)v;
        } else if (MODE == 1) {
          int b = m >> 11, s = m & 2047, h = n >> 6, dk = n & 63;
          ((_Float16*)out)[((size_t)((b * 16 + h) * 64 + dk) << 11) + s] = (_Float16)v;
        } else {
          ((float*)out)[(size_t)m * 1024 + n] = v;
        }
      }
    }
  }
}

// Flash attention: grid (S/128, B*H), 8 waves/block, wave owns 16 query rows.
// K/V^T tiles staged block-wide via TDM (shared by all 8 waves), double buffered.
__global__ __launch_bounds__(256)
void attn_kernel(const _Float16* __restrict__ q,
                 const _Float16* __restrict__ k,
                 const _Float16* __restrict__ vt,
                 _Float16* __restrict__ ctx) {
  __shared__ __align__(16) _Float16 Ksm[2][32 * 64];   // [key][dk]
  __shared__ __align__(16) _Float16 Vsm[2][64 * 32];   // [dv][key]
  __shared__ __align__(16) _Float16 plds[8][16 * 32];  // per-wave P staging

  const int wid  = threadIdx.x >> 5;
  const int lane = threadIdx.x & 31;
  const int bh = blockIdx.y;
  const int b = bh >> 4, h = bh & 15;
  const _Float16* Q  = q  + (size_t)bh * 2048 * 64;
  const _Float16* Kp = k  + (size_t)bh * 2048 * 64;
  const _Float16* Vt = vt + (size_t)bh * 64 * 2048;    // [DK, S]
  const int q0 = blockIdx.x * 128 + wid * 16;
  const int hs = lane >> 4, nl = lane & 15;

  const v16h qa0 = load_frag(Q + (size_t)q0 * 64,      64);
  const v16h qa1 = load_frag(Q + (size_t)q0 * 64 + 32, 64);

  v8f o[4] = {};
  float mr[8], lr[8];
  #pragma unroll
  for (int r = 0; r < 8; r++) { mr[r] = -1e30f; lr[r] = 0.0f; }
  const float scale = 0.125f;

  if (threadIdx.x < 32) {
    tdm_load_2d(lds_offset(&Ksm[0][0]), Kp,  64, 2048, 64, 32, 64);
    tdm_load_2d(lds_offset(&Vsm[0][0]), Vt, 2048,  64, 32, 64, 2048);
  }

  for (int t = 0; t < 64; ++t) {
    const int buf = t & 1;
    if (threadIdx.x < 32) {
      if (t + 1 < 64) {
        const int kn = (t + 1) * 32;
        tdm_load_2d(lds_offset(&Ksm[buf ^ 1][0]), Kp + (size_t)kn * 64,
                    64, 2048, 64, 32, 64);
        tdm_load_2d(lds_offset(&Vsm[buf ^ 1][0]), Vt + kn,
                    2048, 64, 32, 64, 2048);
        __builtin_amdgcn_s_wait_tensorcnt(2);
      } else {
        __builtin_amdgcn_s_wait_tensorcnt(0);
      }
    }
    __syncthreads();

    // scores: S(16x32) = Q(16x64) . K^T from LDS
    v8f sj[2] = {};
    #pragma unroll
    for (int j = 0; j < 2; j++) {
      v16h b0 = load_frag(&Ksm[buf][(j * 16) * 64],      64);
      v16h b1 = load_frag(&Ksm[buf][(j * 16) * 64 + 32], 64);
      sj[j] = wmma16(qa0, b0, sj[j]);
      sj[j] = wmma16(qa1, b1, sj[j]);
    }

    // online softmax: row r lives in VGPR r across one 16-lane half
    float alpha[8];
    #pragma unroll
    for (int r = 0; r < 8; r++) {
      float tm = fmaxf(sj[0][r], sj[1][r]) * scale;
      #pragma unroll
      for (int off = 1; off < 16; off <<= 1)
        tm = fmaxf(tm, __shfl_xor(tm, off, 32));
      float mn = fmaxf(mr[r], tm);
      alpha[r] = __expf(mr[r] - mn);
      mr[r] = mn;
      lr[r] *= alpha[r];
    }
    #pragma unroll
    for (int d = 0; d < 4; d++)
      #pragma unroll
      for (int r = 0; r < 8; r++)
        o[d][r] = o[d][r] * alpha[r];

    float ps[8];
    #pragma unroll
    for (int r = 0; r < 8; r++) ps[r] = 0.0f;
    #pragma unroll
    for (int j = 0; j < 2; j++) {
      #pragma unroll
      for (int r = 0; r < 8; r++) {
        float p = __expf(sj[j][r] * scale - mr[r]);
        ps[r] += p;
        plds[wid][(hs * 8 + r) * 32 + j * 16 + nl] = (_Float16)p;
      }
    }
    #pragma unroll
    for (int r = 0; r < 8; r++) {
      float tsum = ps[r];
      #pragma unroll
      for (int off = 1; off < 16; off <<= 1)
        tsum += __shfl_xor(tsum, off, 32);
      lr[r] += tsum;
    }

    // Re-layout P (C-tile) -> A-fragment via per-wave LDS (same-wave DS in-order).
    __asm__ volatile("s_wait_dscnt 0x0" ::: "memory");
    const _Float16* pp = &plds[wid][nl * 32 + (hs << 3)];
    v8h lo = *(const v8h*)(pp);
    v8h hi = *(const v8h*)(pp + 16);
    v16h pa = __builtin_shufflevector(lo, hi, 0,1,2,3,4,5,6,7,8,9,10,11,12,13,14,15);

    // O(16x64) += P(16x32) . V(32x64) from LDS V^T tile
    #pragma unroll
    for (int d = 0; d < 4; d++) {
      v16h vb = load_frag(&Vsm[buf][(d * 16) * 32], 32);
      o[d] = wmma16(pa, vb, o[d]);
    }
    __syncthreads();   // tile reads done before refill at t+2
  }

  #pragma unroll
  for (int d = 0; d < 4; d++) {
    #pragma unroll
    for (int r = 0; r < 8; r++) {
      float val = o[d][r] / lr[r];
      int s_ = q0 + hs * 8 + r;
      int dc = h * 64 + d * 16 + nl;
      ctx[((size_t)b * 2048 + s_) * 1024 + dc] = (_Float16)val;
    }
  }
}

extern "C" void kernel_launch(void* const* d_in, const int* in_sizes, int n_in,
                              void* d_out, int out_size, void* d_ws, size_t ws_size,
                              hipStream_t stream) {
  const float* query = (const float*)d_in[0];
  const float* key_  = (const float*)d_in[1];
  const float* value = (const float*)d_in[2];
  const float* Wq = (const float*)d_in[3];
  const float* bq = (const float*)d_in[4];
  const float* Wk = (const float*)d_in[5];
  const float* bk = (const float*)d_in[6];
  const float* Wv = (const float*)d_in[7];
  const float* bv = (const float*)d_in[8];
  const float* Wo = (const float*)d_in[9];
  const float* bo = (const float*)d_in[10];

  char* ws = (char*)d_ws;
  const size_t MB = 1024 * 1024;
  _Float16* Xq  = (_Float16*)(ws + 0 * MB);
  _Float16* Xk  = (_Float16*)(ws + 16 * MB);
  _Float16* Xv  = (_Float16*)(ws + 32 * MB);
  _Float16* Wq6 = (_Float16*)(ws + 48 * MB);
  _Float16* Wk6 = (_Float16*)(ws + 50 * MB);
  _Float16* Wv6 = (_Float16*)(ws + 52 * MB);
  _Float16* Wo6 = (_Float16*)(ws + 54 * MB);
  _Float16* Qb  = (_Float16*)(ws + 56 * MB);
  _Float16* Kb  = (_Float16*)(ws + 72 * MB);
  _Float16* Vb  = (_Float16*)(ws + 88 * MB);
  _Float16* Ctx = (_Float16*)(ws + 104 * MB);

  const int nAct = 4 * 2048 * 1024;
  const int nW   = 1024 * 1024;
  dim3 cblk(256);
  cvt_f32_f16_kernel<<<(nAct + 255) / 256, cblk, 0, stream>>>(query, Xq, nAct);
  cvt_f32_f16_kernel<<<(nAct + 255) / 256, cblk, 0, stream>>>(key_,  Xk, nAct);
  cvt_f32_f16_kernel<<<(nAct + 255) / 256, cblk, 0, stream>>>(value, Xv, nAct);
  cvt_f32_f16_kernel<<<(nW + 255) / 256, cblk, 0, stream>>>(Wq, Wq6, nW);
  cvt_f32_f16_kernel<<<(nW + 255) / 256, cblk, 0, stream>>>(Wk, Wk6, nW);
  cvt_f32_f16_kernel<<<(nW + 255) / 256, cblk, 0, stream>>>(Wv, Wv6, nW);
  cvt_f32_f16_kernel<<<(nW + 255) / 256, cblk, 0, stream>>>(Wo, Wo6, nW);

  dim3 ggrid(1024 / 128, 8192 / 128);
  dim3 gblk(256);
  gemm_wmma_kernel<0><<<ggrid, gblk, 0, stream>>>(Xq, Wq6, bq, (void*)Qb);
  gemm_wmma_kernel<0><<<ggrid, gblk, 0, stream>>>(Xk, Wk6, bk, (void*)Kb);
  gemm_wmma_kernel<1><<<ggrid, gblk, 0, stream>>>(Xv, Wv6, bv, (void*)Vb);

  attn_kernel<<<dim3(2048 / 128, 64), gblk, 0, stream>>>(Qb, Kb, Vb, Ctx);

  gemm_wmma_kernel<2><<<ggrid, gblk, 0, stream>>>(Ctx, Wo6, bo, d_out);
}
